// box_tower_82669530513453
// MI455X (gfx1250) — matrix operand order repaired and checked by reading, not compiled, not run on hardware
//
#include <hip/hip_runtime.h>
#include <stdint.h>

// ---------------------------------------------------------------------------
// Types for CDNA5 WMMA (wave32): bf16 A/B fragments (8 VGPRs), f32 C/D (8 VGPRs)
// ---------------------------------------------------------------------------
typedef __attribute__((ext_vector_type(16))) __bf16 bfvec16;
typedef __attribute__((ext_vector_type(8)))  __bf16 bfvec8;
typedef __attribute__((ext_vector_type(8)))  float  fvec8;

__device__ __forceinline__ uint16_t f32_to_bf16(float f) {
  union { float f; uint32_t u; } x; x.f = f;
  uint32_t u = x.u;
  uint32_t r = (u + 0x7FFFu + ((u >> 16) & 1u)) >> 16;   // round-to-nearest-even
  return (uint16_t)r;
}
__device__ __forceinline__ float bf16_to_f32(uint16_t h) {
  union { uint32_t u; float f; } x; x.u = ((uint32_t)h) << 16;
  return x.f;
}

// ---------------------------------------------------------------------------
// NCHW f32 -> NHWC bf16 (B=64, C=256)
// ---------------------------------------------------------------------------
__global__ void k_nchw_to_nhwc_bf16(const float* __restrict__ in,
                                    uint16_t* __restrict__ out,
                                    int H, int W) {
  long total = 64L * 256 * H * W;
  long i = (long)blockIdx.x * blockDim.x + threadIdx.x;
  if (i >= total) return;
  int x = (int)(i % W); long t = i / W;
  int y = (int)(t % H); t /= H;
  int c = (int)(t % 256); int b = (int)(t / 256);
  out[(((long)b * H + y) * W + x) * 256 + c] = f32_to_bf16(in[i]);
}

// ---------------------------------------------------------------------------
// Fold BN scale into conv weights, transpose OIHW -> [tap][oc][cin] bf16.
// shift[oc] = beta - mean*scale  (added in conv epilogue)
// ---------------------------------------------------------------------------
__global__ void k_fold_weights(const float* __restrict__ w,   // [256][256][3][3]
                               const float* __restrict__ g,
                               const float* __restrict__ bb,
                               const float* __restrict__ m,
                               const float* __restrict__ v,
                               uint16_t* __restrict__ wt,     // [9][256][256]
                               float* __restrict__ shift) {   // [256]
  long i = (long)blockIdx.x * blockDim.x + threadIdx.x;
  const long total = 9L * 256 * 256;
  if (i >= total) return;
  int cin = (int)(i % 256); long t2 = i / 256;
  int oc = (int)(t2 % 256); int t = (int)(t2 / 256);
  float scale = g[oc] * rsqrtf(v[oc] + 1e-5f);
  float wv = w[(long)(oc * 256 + cin) * 9 + t] * scale;
  wt[i] = f32_to_bf16(wv);
  if (i < 256) {
    float sc = g[i] * rsqrtf(v[i] + 1e-5f);
    shift[i] = bb[i] - m[i] * sc;
  }
}

// ---------------------------------------------------------------------------
// Implicit-GEMM conv via v_wmma_f32_16x16x32_bf16, register-blocked:
// each wave computes a 16(oc) x 64(pos) tile = 4 accumulators; the A (weight)
// fragment is loaded once per K-chunk and reused by 4 WMMAs.
//   A (16x32, M=oc, K=cin chunk)  : weights  [tap][oc][cin] bf16
//   B (32x16, K=cin chunk, N=pos) : activations NHWC bf16
//   D (16x16 f32)                 : lane = N(pos), vgpr r = M row (split @ lane16)
// Epilogue: + BN shift, ReLU, store NCHW with pitch `opitch` (bf16 or f32).
// Block = 128 threads = 4 waves; wave handles one 16-oc tile.
// ---------------------------------------------------------------------------
template <bool OUT_F32, bool PADDED>
__global__ void k_conv_wmma(const uint16_t* __restrict__ act,  // NHWC bf16 [64][Hi][Wi][256]
                            const uint16_t* __restrict__ wt,   // [9][256][256] bf16 (BN-folded)
                            const float* __restrict__ shift,   // [256]
                            void* __restrict__ outv,           // NCHW [64][256][opitch]
                            int Hi, int Wi, int Ho, int Wo,
                            int dilh, int dilw, int opitch) {
  const int lane = threadIdx.x & 31;
  const int wave = threadIdx.x >> 5;
  const int b    = blockIdx.z;
  const int oc0  = (blockIdx.y * 4 + wave) * 16;
  const int npos = Ho * Wo;
  const int p0   = blockIdx.x * 64;
  const int ln   = lane & 15;
  const int kh   = lane >> 4;

  int  oyv[4], oxv[4];
  bool pval[4];
#pragma unroll
  for (int s = 0; s < 4; ++s) {
    const int p = p0 + s * 16 + ln;
    pval[s] = (p < npos);
    const int pc = pval[s] ? p : (npos - 1);
    oyv[s] = pc / Wo;
    oxv[s] = pc % Wo;
  }

  fvec8 acc[4];
#pragma unroll
  for (int s = 0; s < 4; ++s) acc[s] = (fvec8){};

  const uint16_t* actb = act + (long)b * Hi * Wi * 256;

  for (int t = 0; t < 9; ++t) {
    const int ty = t / 3, tx = t % 3;
    const uint16_t* arow[4];
    bool inb[4];
#pragma unroll
    for (int s = 0; s < 4; ++s) {
      int iy = oyv[s] + ty * dilh;
      int ix = oxv[s] + tx * dilw;
      inb[s] = true;
      if (PADDED) {
        iy -= 1; ix -= 1;
        inb[s] = (iy >= 0) && (iy < Hi) && (ix >= 0) && (ix < Wi);
        if (!inb[s]) { iy = 0; ix = 0; }
      }
      arow[s] = actb + ((long)iy * Wi + ix) * 256 + kh * 16;
      __builtin_prefetch((const void*)arow[s], 0, 0);  // global_prefetch_b8
    }
    const uint16_t* wrow = wt + (long)(t * 256 + oc0 + ln) * 256 + kh * 16;
#pragma unroll
    for (int c0 = 0; c0 < 256; c0 += 32) {
      // A fragment: 16 consecutive cin (k = 16*half + s, matches B), reused 4x
      bfvec8 wlo = *(const bfvec8*)(wrow + c0);
      bfvec8 whi = *(const bfvec8*)(wrow + c0 + 8);
      bfvec16 afrag = __builtin_shufflevector(
          wlo, whi, 0, 1, 2, 3, 4, 5, 6, 7, 8, 9, 10, 11, 12, 13, 14, 15);
#pragma unroll
      for (int s = 0; s < 4; ++s) {
        // B fragment: 16 consecutive cin (32B aligned)
        bfvec16 bfrag = {};
        if (!PADDED || inb[s]) bfrag = *(const bfvec16*)(arow[s] + c0);
        acc[s] = __builtin_amdgcn_wmma_f32_16x16x32_bf16(
            false, afrag, false, bfrag, (short)0, acc[s], false, false);
      }
    }
  }

#pragma unroll
  for (int s = 0; s < 4; ++s) {
    const int p = p0 + s * 16 + ln;
#pragma unroll
    for (int r = 0; r < 8; ++r) {
      const int oc = oc0 + kh * 8 + r;          // C/D layout: M = r (+8 for lanes 16-31)
      float val = acc[s][r] + shift[oc];
      val = fmaxf(val, 0.0f);                    // ReLU
      if (pval[s]) {
        const long oidx = ((long)b * 256 + oc) * opitch + p;
        if constexpr (OUT_F32) ((float*)outv)[oidx] = val;
        else                   ((uint16_t*)outv)[oidx] = f32_to_bf16(val);
      }
    }
  }
}

// ---------------------------------------------------------------------------
// Depthwise cross-correlation of 3 (x,z) pairs + softmax-weighted sum.
// One block per (b,c).  Tiles are staged into LDS with CDNA5 async copies
// (global_load_async_to_lds_b128 + s_wait_asynccnt), kept as bf16 in LDS.
// x planes (pitch-padded): 848(29x29), 784(27x29), 784(29x27)
// z planes (pitch-padded): 176(13x13), 144(11x13), 144(13x11); out 17x17.
// Outputs NCHW f32 and NHWC bf16 (for tower conv).
// ---------------------------------------------------------------------------
__global__ void k_xcorr3(const uint16_t* __restrict__ x0, const uint16_t* __restrict__ x1,
                         const uint16_t* __restrict__ x2,
                         const uint16_t* __restrict__ z0, const uint16_t* __restrict__ z1,
                         const uint16_t* __restrict__ z2,
                         const float* __restrict__ w3,
                         float* __restrict__ out_nchw,      // [64][256][289]
                         uint16_t* __restrict__ out_nhwc) { // [64][17][17][256]
  __shared__ uint16_t s_x[2416];   // 848 + 784 + 784 (bf16)
  __shared__ uint16_t s_z[464];    // 176 + 144 + 144 (bf16)
  const int bc = blockIdx.x, b = bc >> 8, c = bc & 255;
  const int tid = threadIdx.x;

  const uint16_t* gx[3] = { x0 + (long)bc * 848, x1 + (long)bc * 784, x2 + (long)bc * 784 };
  const uint16_t* gz[3] = { z0 + (long)bc * 176, z1 + (long)bc * 144, z2 + (long)bc * 144 };
  const int xchunks[3] = {106, 98, 98};   // 16B chunks per plane
  const int zchunks[3] = {22, 18, 18};
  const int xbase[3]   = {0, 848, 1632};
  const int zbase[3]   = {0, 176, 320};

#pragma unroll
  for (int q = 0; q < 3; ++q) {
    if (tid < xchunks[q]) {
      uint32_t lds = (uint32_t)(uintptr_t)(&s_x[xbase[q] + tid * 8]);
      uint64_t ga  = (uint64_t)(uintptr_t)(gx[q] + tid * 8);
      asm volatile("global_load_async_to_lds_b128 %0, %1, off"
                   :: "v"(lds), "v"(ga) : "memory");
    }
    if (tid < zchunks[q]) {
      uint32_t lds = (uint32_t)(uintptr_t)(&s_z[zbase[q] + tid * 8]);
      uint64_t ga  = (uint64_t)(uintptr_t)(gz[q] + tid * 8);
      asm volatile("global_load_async_to_lds_b128 %0, %1, off"
                   :: "v"(lds), "v"(ga) : "memory");
    }
  }
  asm volatile("s_wait_asynccnt 0x0" ::: "memory");
  __syncthreads();

  const float a0 = w3[0], a1 = w3[1], a2 = w3[2];
  const float mx = fmaxf(a0, fmaxf(a1, a2));
  const float e0 = __expf(a0 - mx), e1 = __expf(a1 - mx), e2 = __expf(a2 - mx);
  const float inv = 1.0f / (e0 + e1 + e2);
  const float wt0 = e0 * inv, wt1 = e1 * inv, wt2 = e2 * inv;

  for (int o = tid; o < 289; o += 256) {
    const int oy = o / 17, ox = o % 17;
    float acc = 0.0f;
    {
      float s = 0.0f;
      for (int zy = 0; zy < 13; ++zy) {
        const uint16_t* xr = &s_x[(oy + zy) * 29 + ox];
        const uint16_t* zr = &s_z[zy * 13];
        for (int zx = 0; zx < 13; ++zx) s = fmaf(bf16_to_f32(xr[zx]), bf16_to_f32(zr[zx]), s);
      }
      acc += wt0 * s;
    }
    {
      float s = 0.0f;
      for (int zy = 0; zy < 11; ++zy) {
        const uint16_t* xr = &s_x[848 + (oy + zy) * 29 + ox];
        const uint16_t* zr = &s_z[176 + zy * 13];
        for (int zx = 0; zx < 13; ++zx) s = fmaf(bf16_to_f32(xr[zx]), bf16_to_f32(zr[zx]), s);
      }
      acc += wt1 * s;
    }
    {
      float s = 0.0f;
      for (int zy = 0; zy < 13; ++zy) {
        const uint16_t* xr = &s_x[1632 + (oy + zy) * 27 + ox];
        const uint16_t* zr = &s_z[320 + zy * 11];
        for (int zx = 0; zx < 11; ++zx) s = fmaf(bf16_to_f32(xr[zx]), bf16_to_f32(zr[zx]), s);
      }
      acc += wt2 * s;
    }
    out_nchw[(long)bc * 289 + o] = acc;
    out_nhwc[(((long)b * 17 + oy) * 17 + ox) * 256 + c] = f32_to_bf16(acc);
  }
}

// ---------------------------------------------------------------------------
// bbox pred: conv 256->4 (3x3 pad1) + exp(min(adjust*y + bias, 4.7))
// one block per (b, oc)
// ---------------------------------------------------------------------------
__global__ void k_bbox_pred(const float* __restrict__ xreg,   // [64][256][289]
                            const float* __restrict__ bpw,    // [4][256][3][3]
                            const float* __restrict__ bpb,    // [4]
                            const float* __restrict__ adjust, // [1]
                            const float* __restrict__ bias,   // [4]
                            float* __restrict__ out) {        // [64][4][289]
  __shared__ float sw[2304];
  const int b = blockIdx.x >> 2, oc = blockIdx.x & 3;
  for (int i = threadIdx.x; i < 2304; i += blockDim.x) sw[i] = bpw[(long)oc * 2304 + i];
  __syncthreads();
  const float adj = adjust[0];
  const float bs = bias[oc];
  for (int o = threadIdx.x; o < 289; o += blockDim.x) {
    const int oy = o / 17, ox = o % 17;
    float acc = bpb[oc];
    for (int c = 0; c < 256; ++c) {
      const float* xc = xreg + ((long)b * 256 + c) * 289;
      const float* wc = sw + c * 9;
#pragma unroll
      for (int ty = 0; ty < 3; ++ty) {
        const int iy = oy + ty - 1;
        if ((unsigned)iy >= 17u) continue;
#pragma unroll
        for (int tx = 0; tx < 3; ++tx) {
          const int ix = ox + tx - 1;
          if ((unsigned)ix >= 17u) continue;
          acc = fmaf(xc[iy * 17 + ix], wc[ty * 3 + tx], acc);
        }
      }
    }
    const float v = fminf(adj * acc + bs, 4.7f);
    out[((long)b * 4 + oc) * 289 + o] = __expf(v);
  }
}

// ---------------------------------------------------------------------------
// cls pred: conv 256->1 (3x3 pad1), * 0.1.  one block per b
// ---------------------------------------------------------------------------
__global__ void k_cls_pred(const float* __restrict__ cin,  // [64][256][289]
                           const float* __restrict__ cpw,  // [1][256][3][3]
                           const float* __restrict__ cpb,  // [1]
                           float* __restrict__ out) {      // [64][289]
  __shared__ float sw[2304];
  const int b = blockIdx.x;
  for (int i = threadIdx.x; i < 2304; i += blockDim.x) sw[i] = cpw[i];
  __syncthreads();
  for (int o = threadIdx.x; o < 289; o += blockDim.x) {
    const int oy = o / 17, ox = o % 17;
    float acc = cpb[0];
    for (int c = 0; c < 256; ++c) {
      const float* xc = cin + ((long)b * 256 + c) * 289;
      const float* wc = sw + c * 9;
#pragma unroll
      for (int ty = 0; ty < 3; ++ty) {
        const int iy = oy + ty - 1;
        if ((unsigned)iy >= 17u) continue;
#pragma unroll
        for (int tx = 0; tx < 3; ++tx) {
          const int ix = ox + tx - 1;
          if ((unsigned)ix >= 17u) continue;
          acc = fmaf(xc[iy * 17 + ix], wc[ty * 3 + tx], acc);
        }
      }
    }
    out[(long)b * 289 + o] = 0.1f * acc;
  }
}

// ---------------------------------------------------------------------------
// Host driver
// ---------------------------------------------------------------------------
extern "C" void kernel_launch(void* const* d_in, const int* in_sizes, int n_in,
                              void* d_out, int out_size, void* d_ws, size_t ws_size,
                              hipStream_t stream) {
  (void)in_sizes; (void)n_in; (void)out_size; (void)ws_size;
  const float* search = (const float*)d_in[0];
  const float* kern   = (const float*)d_in[1];
  const float* mset[2][5] = {
      {(const float*)d_in[2], (const float*)d_in[3], (const float*)d_in[4],
       (const float*)d_in[5], (const float*)d_in[6]},
      {(const float*)d_in[7], (const float*)d_in[8], (const float*)d_in[9],
       (const float*)d_in[10], (const float*)d_in[11]}};
  const float* cls_dw_w = (const float*)d_in[12];
  const float* reg_dw_w = (const float*)d_in[13];
  const float* btw = (const float*)d_in[14];
  const float* btg = (const float*)d_in[15];
  const float* btb = (const float*)d_in[16];
  const float* btm = (const float*)d_in[17];
  const float* btv = (const float*)d_in[18];
  const float* ctw = (const float*)d_in[19];
  const float* ctg = (const float*)d_in[20];
  const float* ctb = (const float*)d_in[21];
  const float* ctm = (const float*)d_in[22];
  const float* ctv = (const float*)d_in[23];
  const float* bpw = (const float*)d_in[24];
  const float* bpb = (const float*)d_in[25];
  const float* cpw = (const float*)d_in[26];
  const float* cpb = (const float*)d_in[27];
  const float* adjust = (const float*)d_in[28];
  const float* bias = (const float*)d_in[29];

  // ---- workspace carve (deterministic) ----
  char* ws = (char*)d_ws;
  size_t off = 0;
  auto carve = [&](size_t bytes) -> char* {
    off = (off + 255) & ~(size_t)255;
    char* p = ws + off;
    off += bytes;
    return p;
  };
  uint16_t* searchN = (uint16_t*)carve(64L * 31 * 31 * 256 * 2);
  uint16_t* kernN   = (uint16_t*)carve(64L * 15 * 15 * 256 * 2);
  uint16_t* wts     = (uint16_t*)carve(14L * 589824 * 2);
  float*    shifts  = (float*)carve(14L * 256 * 4);
  // pitch-padded bf16 NCHW conv outputs (16B-aligned planes for async b128)
  const int zpit[3] = {176, 144, 144};   // for 169, 143, 143
  const int xpit[3] = {848, 784, 784};   // for 841, 783, 783
  uint16_t* zout[2][3];
  uint16_t* xout[2][3];
  for (int br = 0; br < 2; ++br)
    for (int j = 0; j < 3; ++j) {
      zout[br][j] = (uint16_t*)carve(64L * 256 * zpit[j] * 2);
      xout[br][j] = (uint16_t*)carve(64L * 256 * xpit[j] * 2);
    }
  float*    reg_dw  = (float*)carve(4734976L * 4);
  uint16_t* cls_dwN = (uint16_t*)carve(4734976L * 2);
  uint16_t* reg_dwN = (uint16_t*)carve(4734976L * 2);
  float*    cls_tw  = (float*)carve(4734976L * 4);

  float* outF = (float*)d_out;
  float* out_x     = outF;                 // [64][4][289]
  float* out_cls   = outF + 73984;         // [64][1][289]
  float* out_clsdw = outF + 92480;         // [64][256][289]
  float* out_xreg  = outF + 4827456;       // [64][256][289]

  // 1) activations -> NHWC bf16
  {
    long t = 64L * 256 * 31 * 31;
    k_nchw_to_nhwc_bf16<<<(unsigned)((t + 255) / 256), 256, 0, stream>>>(search, searchN, 31, 31);
  }
  {
    long t = 64L * 256 * 15 * 15;
    k_nchw_to_nhwc_bf16<<<(unsigned)((t + 255) / 256), 256, 0, stream>>>(kern, kernN, 15, 15);
  }

  // 2) fold BN into weights (12 matrix convs + 2 towers)
  for (int br = 0; br < 2; ++br)
    for (int i = 0; i < 6; ++i) {
      const int widx = br * 6 + i;
      k_fold_weights<<<2304, 256, 0, stream>>>(
          mset[br][0] + (long)i * 256 * 256 * 9,
          mset[br][1] + i * 256, mset[br][2] + i * 256,
          mset[br][3] + i * 256, mset[br][4] + i * 256,
          wts + (long)widx * 589824, shifts + widx * 256);
    }
  k_fold_weights<<<2304, 256, 0, stream>>>(btw, btg, btb, btm, btv,
                                           wts + 12L * 589824, shifts + 12 * 256);
  k_fold_weights<<<2304, 256, 0, stream>>>(ctw, ctg, ctb, ctm, ctv,
                                           wts + 13L * 589824, shifts + 13 * 256);

  // 3) matrix convs (WMMA). pair j: z conv idx 2j, x conv idx 2j+1, shared dilation
  const int dH[3] = {1, 2, 1}, dW[3] = {1, 1, 2};
  const int zH[3] = {13, 11, 13}, zW[3] = {13, 13, 11};
  const int xH[3] = {29, 27, 29}, xW[3] = {29, 29, 27};
  for (int br = 0; br < 2; ++br)
    for (int j = 0; j < 3; ++j) {
      const int wz = br * 6 + 2 * j;
      const int wx = br * 6 + 2 * j + 1;
      {
        const int np = zH[j] * zW[j];
        dim3 g((np + 63) / 64, 4, 64);
        k_conv_wmma<false, false><<<g, 128, 0, stream>>>(
            kernN, wts + (long)wz * 589824, shifts + wz * 256, zout[br][j],
            15, 15, zH[j], zW[j], dH[j], dW[j], zpit[j]);
      }
      {
        const int np = xH[j] * xW[j];
        dim3 g((np + 63) / 64, 4, 64);
        k_conv_wmma<false, false><<<g, 128, 0, stream>>>(
            searchN, wts + (long)wx * 589824, shifts + wx * 256, xout[br][j],
            31, 31, xH[j], xW[j], dH[j], dW[j], xpit[j]);
      }
    }

  // 4) depthwise xcorr + softmax-weighted sum (async-to-LDS staging)
  k_xcorr3<<<16384, 256, 0, stream>>>(xout[0][0], xout[0][1], xout[0][2],
                                      zout[0][0], zout[0][1], zout[0][2],
                                      cls_dw_w, out_clsdw, cls_dwN);
  k_xcorr3<<<16384, 256, 0, stream>>>(xout[1][0], xout[1][1], xout[1][2],
                                      zout[1][0], zout[1][1], zout[1][2],
                                      reg_dw_w, reg_dw, reg_dwN);

  // 5) tower convs (WMMA, pad 1, f32 NCHW output, unpadded pitch 289)
  {
    dim3 g((289 + 63) / 64, 4, 64);
    k_conv_wmma<true, true><<<g, 128, 0, stream>>>(
        reg_dwN, wts + 12L * 589824, shifts + 12 * 256, out_xreg, 17, 17, 17, 17, 1, 1, 289);
    k_conv_wmma<true, true><<<g, 128, 0, stream>>>(
        cls_dwN, wts + 13L * 589824, shifts + 13 * 256, cls_tw, 17, 17, 17, 17, 1, 1, 289);
  }

  // 6) prediction heads + elementwise epilogues
  k_bbox_pred<<<256, 320, 0, stream>>>(out_xreg, bpw, bpb, adjust, bias, out_x);
  k_cls_pred<<<64, 320, 0, stream>>>(cls_tw, cpw, cpb, out_cls);
}
